// MultiHeadAttention_14843406975282
// MI455X (gfx1250) — matrix operand order repaired and checked by reading, not compiled, not run on hardware
//
#include <hip/hip_runtime.h>
#include <hip/hip_bf16.h>
#include <math.h>
#include <stdint.h>

// ---------------------------------------------------------------------------
// MI455X (gfx1250) attention block.
// Compute-bound (~47 GFLOP vs ~2us HBM time at 23.3 TB/s) -> every GEMM runs
// on v_wmma_f32_16x16x32_f16 (wave32). Tile movement uses the CDNA5 async
// global->LDS path (ASYNCcnt) so the k-loop double-buffers without VGPR
// staging.
// ---------------------------------------------------------------------------

typedef _Float16 half_t;
typedef __attribute__((ext_vector_type(16))) _Float16 v16h;
typedef __attribute__((ext_vector_type(8)))  _Float16 v8h;
typedef __attribute__((ext_vector_type(4)))  _Float16 v4h;
typedef __attribute__((ext_vector_type(8)))  float    v8f;

union frag16 { v16h v; v8h h[2]; };

#define WMMA_F32_F16(a, b, c) \
    __builtin_amdgcn_wmma_f32_16x16x32_f16(false, (a), false, (b), (short)0, (c), false, false)

// Async copy of 16 bytes from global to LDS (tracked by ASYNCcnt).
__device__ inline void async_copy_b128(const void* gptr, const void* lds_ptr) {
    const uint32_t lds_off = (uint32_t)(uintptr_t)lds_ptr;   // addr[31:0] = LDS offset
    const uint64_t ga      = (uint64_t)(uintptr_t)gptr;
    asm volatile("global_load_async_to_lds_b128 %0, %1, off"
                 :: "v"(lds_off), "v"(ga) : "memory");
}
__device__ inline void wait_async0() {
    asm volatile("s_wait_asynccnt 0x0" ::: "memory");
}

// A-fragment (16x32 f16, MxK): lane half = lane>>4, row = lane&15.
// elems 0..7 -> K = k0 + half*8 + i ; elems 8..15 -> K = k0 + 16 + half*8 + i
__device__ inline v16h load_a_frag(const half_t* __restrict__ p) {
    frag16 f;
    f.h[0] = *(const v8h*)(p);
    f.h[1] = *(const v8h*)(p + 16);
    return f.v;
}
// B-fragment (32x16 f16, KxN) from N-major tile: col = lane&15,
// elems 0..7 -> K = k0 + half*8 + i ; elems 8..15 -> K = k0 + 16 + half*8 + i
__device__ inline v16h load_b_frag(const half_t* __restrict__ p) {
    frag16 f;
    f.h[0] = *(const v8h*)(p);
    f.h[1] = *(const v8h*)(p + 16);
    return f.v;
}

// ---------------------------------------------------------------------------
// f32 -> f16 conversion
// ---------------------------------------------------------------------------
__global__ __launch_bounds__(256)
void cvt_f32_f16_kernel(const float* __restrict__ x, half_t* __restrict__ y, int n) {
    int i = (blockIdx.x * 256 + threadIdx.x) * 4;
    if (i + 3 < n) {
        float4 v = *(const float4*)(x + i);
        y[i + 0] = (half_t)v.x;
        y[i + 1] = (half_t)v.y;
        y[i + 2] = (half_t)v.z;
        y[i + 3] = (half_t)v.w;
    }
}

// ---------------------------------------------------------------------------
// Y = act(X @ W^T + b).  X: MxK f16, W16: NxK f16, Y: MxN (f16 / f32).
// 128x128x32 tiles, 8 waves (wave = 32x64 = 2x4 WMMA tiles), async
// double-buffered LDS staging.
// ---------------------------------------------------------------------------
#define GBM 128
#define GBN 128
#define GBK 32
#define GLDA 40   // f16 stride (32 + 8 pad): 80B rows, 16B aligned
#define GLDB 40

template<bool RELU, bool WH, bool WF>
__global__ __launch_bounds__(256)
void gemm_xwT_bias_act(const half_t* __restrict__ X, const half_t* __restrict__ W16,
                       const float* __restrict__ bias, half_t* __restrict__ Yh,
                       float* __restrict__ Yf, int M, int N, int K) {
    __shared__ half_t As[2][GBM * GLDA];
    __shared__ half_t Bs[2][GBN * GLDB];

    const int tid   = threadIdx.x;
    const int lane  = tid & 31;
    const int wave  = tid >> 5;
    const int half_ = lane >> 4;
    const int l15   = lane & 15;
    const int wr    = wave >> 1;              // 0..3 : 32-row strip
    const int wc    = wave & 1;               // 0..1 : 64-col strip
    const int m0    = blockIdx.x * GBM;
    const int n0    = blockIdx.y * GBN;

    v8f acc[2][4];
    const v8f vzero = {};
#pragma unroll
    for (int i = 0; i < 2; ++i)
#pragma unroll
        for (int j = 0; j < 4; ++j) acc[i][j] = vzero;

    const int ldrow = tid >> 1;               // 0..127
    const int ldsegb = (tid & 1) * 2;         // {0,2} -> two b128 segs of 8 f16

    auto issue_tile = [&](int k0, int buf) {
        const half_t* xs = X   + (size_t)(m0 + ldrow) * K + k0;
        const half_t* wsrc = W16 + (size_t)(n0 + ldrow) * K + k0;
#pragma unroll
        for (int s = 0; s < 2; ++s) {
            const int seg = ldsegb + s;       // 0..3, 8 f16 each
            async_copy_b128(xs + seg * 8,   &As[buf][ldrow * GLDA + seg * 8]);
            async_copy_b128(wsrc + seg * 8, &Bs[buf][ldrow * GLDB + seg * 8]);
        }
    };

    issue_tile(0, 0);
    wait_async0();
    __syncthreads();

    const int nk = K / GBK;
    for (int ki = 0; ki < nk; ++ki) {
        const int cur = ki & 1;
        if (ki + 1 < nk) issue_tile((ki + 1) * GBK, cur ^ 1);

        v16h a0 = load_a_frag(&As[cur][(wr * 32 + 0  + l15) * GLDA + half_ * 8]);
        v16h a1 = load_a_frag(&As[cur][(wr * 32 + 16 + l15) * GLDA + half_ * 8]);
#pragma unroll
        for (int wn = 0; wn < 4; ++wn) {
            v16h b = load_b_frag(&Bs[cur][(wc * 64 + wn * 16 + l15) * GLDB + half_ * 8]);
            acc[0][wn] = WMMA_F32_F16(a0, b, acc[0][wn]);
            acc[1][wn] = WMMA_F32_F16(a1, b, acc[1][wn]);
        }
        wait_async0();
        __syncthreads();
    }

    // epilogue: bias + (relu) + store (compile-time specialized)
#pragma unroll
    for (int wm = 0; wm < 2; ++wm)
#pragma unroll
        for (int wn = 0; wn < 4; ++wn) {
            const int gcol = n0 + wc * 64 + wn * 16 + l15;
            const float bv = bias[gcol];
#pragma unroll
            for (int r = 0; r < 8; ++r) {
                const int grow = m0 + wr * 32 + wm * 16 + r + half_ * 8;
                float val = acc[wm][wn][r] + bv;
                if constexpr (RELU) val = fmaxf(val, 0.0f);
                if constexpr (WH) Yh[(size_t)grow * N + gcol] = (half_t)val;
                if constexpr (WF) Yf[(size_t)grow * N + gcol] = val;
            }
        }
}

// ---------------------------------------------------------------------------
// q_att[s] = sum_d qq[s,d] * qk[s,d]   (one wave32 per row)
// ---------------------------------------------------------------------------
__global__ __launch_bounds__(256)
void qatt_kernel(const half_t* __restrict__ qq, const half_t* __restrict__ qk,
                 float* __restrict__ qatt, int S, int F) {
    const int row  = blockIdx.x * 8 + (threadIdx.x >> 5);
    const int lane = threadIdx.x & 31;
    if (row >= S) return;
    const half_t* a = qq + (size_t)row * F;
    const half_t* b = qk + (size_t)row * F;
    float sum = 0.0f;
    for (int c = lane; c < F / 8; c += 32) {
        v8h av = *(const v8h*)(a + c * 8);
        v8h bv = *(const v8h*)(b + c * 8);
#pragma unroll
        for (int j = 0; j < 8; ++j) sum += (float)av[j] * (float)bv[j];
    }
#pragma unroll
    for (int off = 16; off > 0; off >>= 1) sum += __shfl_xor(sum, off, 32);
    if (lane == 0) qatt[row] = sum;
}

// ---------------------------------------------------------------------------
// Flash attention over S keys with extra self-attention logit column.
// Workgroup: 64 query rows, 8 waves. LDS (dynamic, ~229 KB < 320 KB):
//   Q(64x512 f16), K-block(64x512 f16), V^T-block(512x64 f16),
//   P(64x64 f16), scores(64x64 f32), per-row stats.
// Wave(w): row-tile rw=w>>1 (16 rows), col half ch=w&1 (256 of 512 out cols)
//  -> O accumulator 16x256 f32 = 128 VGPRs.
// Q/K tiles arrive via async global->LDS; V is register-transposed with
// packed b64 stores. Softmax runs 4 threads/row with in-wave shuffles.
// Epilogue fuses: res = qv * q_att + O/l  (f16, feeds the final GEMM).
// ---------------------------------------------------------------------------
#define ABR 64
#define ABC 64
#define ALDQ 520
#define ALDK 520
#define ALDV 72
#define ALDP 72
#define ALDS 68

__global__ __launch_bounds__(256)
void attn_kernel(const half_t* __restrict__ qq, const half_t* __restrict__ kk,
                 const half_t* __restrict__ vv, const half_t* __restrict__ qv,
                 const float* __restrict__ qatt, half_t* __restrict__ res,
                 int S, int F, float scale) {
    extern __shared__ char smem_raw[];
    half_t* Qs  = (half_t*)smem_raw;            // 64*520
    half_t* Ks  = Qs + 64 * ALDQ;               // 64*520
    half_t* Vt  = Ks + 64 * ALDK;               // 512*72   (Vt[f][t])
    half_t* Ps  = Vt + 512 * ALDV;              // 64*72
    float*  Sc  = (float*)(Ps + 64 * ALDP);     // 64*68
    float*  m_s = Sc + 64 * ALDS;
    float*  l_s = m_s + 64;
    float*  al_s = l_s + 64;
    float*  qa_s = al_s + 64;

    const int tid   = threadIdx.x;
    const int lane  = tid & 31;
    const int wave  = tid >> 5;
    const int half_ = lane >> 4;
    const int l15   = lane & 15;
    const int rw    = wave >> 1;    // row tile 0..3
    const int ch    = wave & 1;     // column half 0..1
    const int qbase = blockIdx.x * ABR;

    v8f o[16];
    const v8f vzero = {};
#pragma unroll
    for (int i = 0; i < 16; ++i) o[i] = vzero;

    // async-load Q block (64 x 512)
    for (int idx = tid; idx < 64 * 64; idx += 256) {
        const int row = idx >> 6, seg = idx & 63;
        async_copy_b128(qq + (size_t)(qbase + row) * F + seg * 8,
                        &Qs[row * ALDQ + seg * 8]);
    }
    if (tid < 64) {
        const float qa = qatt[qbase + tid];
        qa_s[tid] = qa;
        m_s[tid]  = qa * scale;   // extra softmax column initializes the max
        l_s[tid]  = 1.0f;         // exp(qa*scale - m) == 1
    }
    wait_async0();
    __syncthreads();

    for (int kb = 0; kb < S; kb += ABC) {
        // async K rows (raw byte copies, no VGPR staging)
        for (int idx = tid; idx < 64 * 64; idx += 256) {
            const int row = idx >> 6, seg = idx & 63;
            async_copy_b128(kk + (size_t)(kb + row) * F + seg * 8,
                            &Ks[row * ALDK + seg * 8]);
        }
        // V: register transpose, 4 keys packed per ds_store_b64
        for (int idx = tid; idx < 16 * 64; idx += 256) {
            const int rp  = idx >> 6;     // key group: rows rp*4..rp*4+3
            const int seg = idx & 63;     // 8 features
            const half_t* vb = vv + (size_t)(kb + rp * 4) * F + seg * 8;
            v8h r0 = *(const v8h*)(vb + 0 * F);
            v8h r1 = *(const v8h*)(vb + 1 * F);
            v8h r2 = *(const v8h*)(vb + 2 * F);
            v8h r3 = *(const v8h*)(vb + 3 * F);
#pragma unroll
            for (int j = 0; j < 8; ++j) {
                v4h pk; pk[0] = r0[j]; pk[1] = r1[j]; pk[2] = r2[j]; pk[3] = r3[j];
                *(v4h*)(&Vt[(seg * 8 + j) * ALDV + rp * 4]) = pk;
            }
        }
        // prefetch next key block while this one is consumed
        if (kb + ABC < S) {
            __builtin_prefetch(kk + (size_t)(kb + ABC + (tid & 63)) * F + (tid >> 6) * 128, 0, 1);
            __builtin_prefetch(vv + (size_t)(kb + ABC + (tid & 63)) * F + (tid >> 6) * 128, 0, 1);
        }
        wait_async0();
        __syncthreads();

        // scores: wave computes a 16x32 sub-tile of the 64x64 score block
        const int sc0 = ch * 2;
        v8f s0 = vzero, s1 = vzero;
        for (int d0 = 0; d0 < F; d0 += 32) {
            v16h a  = load_a_frag(&Qs[(rw * 16 + l15) * ALDQ + d0 + half_ * 8]);
            v16h b0 = load_b_frag(&Ks[((sc0 + 0) * 16 + l15) * ALDK + d0 + half_ * 8]);
            v16h b1 = load_b_frag(&Ks[((sc0 + 1) * 16 + l15) * ALDK + d0 + half_ * 8]);
            s0 = WMMA_F32_F16(a, b0, s0);
            s1 = WMMA_F32_F16(a, b1, s1);
        }
#pragma unroll
        for (int r = 0; r < 8; ++r) {
            const int row = rw * 16 + r + half_ * 8;
            Sc[row * ALDS + (sc0 + 0) * 16 + l15] = s0[r] * scale;
            Sc[row * ALDS + (sc0 + 1) * 16 + l15] = s1[r] * scale;
        }
        __syncthreads();

        // online softmax: 4 threads per row, in-wave shuffle combines
        {
            const int srow = tid >> 2;            // 0..63
            const int t4   = tid & 3;             // 16 cols each
            const float mold = m_s[srow];
            float sv[16];
            float pmax = mold;
            const float* scr = &Sc[srow * ALDS + t4 * 16];
#pragma unroll
            for (int j = 0; j < 16; ++j) { sv[j] = scr[j]; pmax = fmaxf(pmax, sv[j]); }
            pmax = fmaxf(pmax, __shfl_xor(pmax, 1, 32));
            pmax = fmaxf(pmax, __shfl_xor(pmax, 2, 32));
            float lpart = 0.0f;
            v8h p0, p1;
#pragma unroll
            for (int j = 0; j < 8; ++j) {
                float pa = __expf(sv[j] - pmax);
                float pb = __expf(sv[j + 8] - pmax);
                lpart += pa + pb;
                p0[j] = (half_t)pa; p1[j] = (half_t)pb;
            }
            *(v8h*)(&Ps[srow * ALDP + t4 * 16 + 0]) = p0;
            *(v8h*)(&Ps[srow * ALDP + t4 * 16 + 8]) = p1;
            lpart += __shfl_xor(lpart, 1, 32);
            lpart += __shfl_xor(lpart, 2, 32);
            if (t4 == 0) {
                const float alpha = __expf(mold - pmax);
                m_s[srow]  = pmax;
                l_s[srow]  = l_s[srow] * alpha + lpart;
                al_s[srow] = alpha;
            }
        }
        __syncthreads();

        // O = O*alpha + P @ V
        float alpha_r[8];
#pragma unroll
        for (int r = 0; r < 8; ++r) alpha_r[r] = al_s[rw * 16 + r + half_ * 8];
#pragma unroll
        for (int ct = 0; ct < 16; ++ct)
#pragma unroll
            for (int r = 0; r < 8; ++r) o[ct][r] *= alpha_r[r];

#pragma unroll
        for (int t0 = 0; t0 < ABC; t0 += 32) {
            v16h a = load_a_frag(&Ps[(rw * 16 + l15) * ALDP + t0 + half_ * 8]);
#pragma unroll
            for (int ct = 0; ct < 16; ++ct) {
                v16h b = load_b_frag(&Vt[(ch * 256 + ct * 16 + l15) * ALDV + t0 + half_ * 8]);
                o[ct] = WMMA_F32_F16(a, b, o[ct]);
            }
        }
        __syncthreads();
    }

    // finalize: res = qv * q_att + O / l
    float linv_r[8], qa_r[8];
#pragma unroll
    for (int r = 0; r < 8; ++r) {
        const int row = rw * 16 + r + half_ * 8;
        linv_r[r] = 1.0f / l_s[row];
        qa_r[r]   = qa_s[row];
    }
#pragma unroll
    for (int ct = 0; ct < 16; ++ct) {
        const int gcol = ch * 256 + ct * 16 + l15;
#pragma unroll
        for (int r = 0; r < 8; ++r) {
            const int grow = qbase + rw * 16 + r + half_ * 8;
            const float val = o[ct][r] * linv_r[r] +
                              qa_r[r] * (float)qv[(size_t)grow * F + gcol];
            res[(size_t)grow * F + gcol] = (half_t)val;
        }
    }
}

// ---------------------------------------------------------------------------
// host launcher
// ---------------------------------------------------------------------------
extern "C" void kernel_launch(void* const* d_in, const int* in_sizes, int n_in,
                              void* d_out, int out_size, void* d_ws, size_t ws_size,
                              hipStream_t stream) {
    const int S = 4096, F = 512;
    const float scale = 1.0f / sqrtf((float)(F + 1));

    const float* q  = (const float*)d_in[0];
    const float* k  = (const float*)d_in[1];
    const float* v  = (const float*)d_in[2];
    const float* Wq = (const float*)d_in[3];
    const float* bq = (const float*)d_in[4];
    const float* Wk = (const float*)d_in[5];
    const float* bk = (const float*)d_in[6];
    const float* Wv = (const float*)d_in[7];
    const float* bv = (const float*)d_in[8];
    const float* Wo = (const float*)d_in[9];
    const float* bo = (const float*)d_in[10];
    float* out = (float*)d_out;

    char* ws = (char*)d_ws;
    const size_t NE  = (size_t)S * F;   // 2M elements
    const size_t FF  = (size_t)F * F;   // 256K elements
    half_t* q16   = (half_t*)(ws + 0 * NE * 2);
    half_t* k16   = (half_t*)(ws + 1 * NE * 2);
    half_t* v16   = (half_t*)(ws + 2 * NE * 2);
    half_t* qq16  = (half_t*)(ws + 3 * NE * 2);
    half_t* kk16  = (half_t*)(ws + 4 * NE * 2);
    half_t* vv16  = (half_t*)(ws + 5 * NE * 2);
    half_t* qk16  = (half_t*)(ws + 6 * NE * 2);
    half_t* qv16  = (half_t*)(ws + 7 * NE * 2);
    half_t* res16 = (half_t*)(ws + 8 * NE * 2);
    float*  qatt  = (float*)(ws + 9 * NE * 2);
    char*   wbase = ws + 9 * NE * 2 + 65536;
    half_t* Wq16  = (half_t*)(wbase + 0 * FF * 2);
    half_t* Wk16  = (half_t*)(wbase + 1 * FF * 2);
    half_t* Wv16  = (half_t*)(wbase + 2 * FF * 2);
    half_t* Wo16  = (half_t*)(wbase + 3 * FF * 2);

    const int cvtA = (int)(NE / 1024);
    const int cvtW = (int)(FF / 1024);
    cvt_f32_f16_kernel<<<cvtA, 256, 0, stream>>>(q, q16, (int)NE);
    cvt_f32_f16_kernel<<<cvtA, 256, 0, stream>>>(k, k16, (int)NE);
    cvt_f32_f16_kernel<<<cvtA, 256, 0, stream>>>(v, v16, (int)NE);
    cvt_f32_f16_kernel<<<cvtW, 256, 0, stream>>>(Wq, Wq16, (int)FF);
    cvt_f32_f16_kernel<<<cvtW, 256, 0, stream>>>(Wk, Wk16, (int)FF);
    cvt_f32_f16_kernel<<<cvtW, 256, 0, stream>>>(Wv, Wv16, (int)FF);
    cvt_f32_f16_kernel<<<cvtW, 256, 0, stream>>>(Wo, Wo16, (int)FF);

    dim3 ggrid(S / GBM, F / GBN);  // (32, 4)
    gemm_xwT_bias_act<true,  true,  false><<<ggrid, 256, 0, stream>>>(q16, Wq16, bq, qq16, nullptr, S, F, F);
    gemm_xwT_bias_act<true,  true,  false><<<ggrid, 256, 0, stream>>>(k16, Wk16, bk, kk16, nullptr, S, F, F);
    gemm_xwT_bias_act<true,  true,  false><<<ggrid, 256, 0, stream>>>(v16, Wv16, bv, vv16, nullptr, S, F, F);
    gemm_xwT_bias_act<false, true,  false><<<ggrid, 256, 0, stream>>>(q16, Wk16, bk, qk16, nullptr, S, F, F);
    gemm_xwT_bias_act<false, true,  false><<<ggrid, 256, 0, stream>>>(q16, Wv16, bv, qv16, nullptr, S, F, F);

    qatt_kernel<<<S / 8, 256, 0, stream>>>(qq16, qk16, qatt, S, F);

    const size_t smem = (size_t)(64 * ALDQ + 64 * ALDK + 512 * ALDV + 64 * ALDP) * 2 +
                        (size_t)(64 * ALDS + 4 * 64) * 4;
    hipFuncSetAttribute((const void*)attn_kernel,
                        hipFuncAttributeMaxDynamicSharedMemorySize, (int)smem);
    attn_kernel<<<S / ABR, 256, smem, stream>>>(qq16, kk16, vv16, qv16, qatt, res16,
                                                S, F, scale);

    gemm_xwT_bias_act<true, false, true><<<ggrid, 256, 0, stream>>>(res16, Wo16, bo, nullptr, out, S, F, F);
}
